// LA_Model_41257455845902
// MI455X (gfx1250) — matrix operand order repaired and checked by reading, not compiled, not run on hardware
//
#include <hip/hip_runtime.h>
#include <math.h>

// ---------------- model dims ----------------
#define B_    128
#define T_    256
#define H_    512
#define E_    300
#define E_PAD 320
#define FEAT_ 61
#define NEG_  0.01f
#define WROW  40   // LDS weight-tile row pitch in bf16 (32 data + 8 pad -> 80B, bank spread)

typedef __attribute__((ext_vector_type(16))) __bf16 bf16x16;
typedef __attribute__((ext_vector_type(8)))  __bf16 bf16x8;
typedef __attribute__((ext_vector_type(8)))  float  f32x8;
typedef __attribute__((ext_vector_type(4)))  unsigned u32x4;
typedef __attribute__((ext_vector_type(8)))  int      i32x8;
typedef __attribute__((ext_vector_type(4)))  int      i32x4;

#if defined(__has_builtin)
#if __has_builtin(__builtin_amdgcn_tensor_load_to_lds) && __has_builtin(__builtin_amdgcn_s_wait_tensorcnt)
#define USE_TDM 1
#endif
#endif

static __device__ __forceinline__ __bf16 f2bf(float f) {
  union { float f; unsigned u; } v; v.f = f;
  unsigned r = v.u + 0x7FFFu + ((v.u >> 16) & 1u);   // round-to-nearest-even
  unsigned short h = (unsigned short)(r >> 16);
  return __builtin_bit_cast(__bf16, h);
}
static __device__ __forceinline__ float bf2f(__bf16 b) {
  unsigned short h = __builtin_bit_cast(unsigned short, b);
  union { unsigned u; float f; } v; v.u = ((unsigned)h) << 16;
  return v.f;
}
static __device__ __forceinline__ float sigf(float x) { return 1.f / (1.f + expf(-x)); }

// ---------------------------------------------------------------------------
// Blocked WMMA bf16 GEMM (M fixed = 128):
//   D[z][m][n] = bias[z][n] + A1[z]W1[z]^T (K1) + A2[z]W2[z]^T (K2)
// Block: 256 threads = 8 waves; wave w owns M-tile w (16 rows) x 64 columns.
// Weight chunk (64 N-rows x 32 K) staged to LDS once per chunk via the Tensor
// Data Mover (double buffered, s_wait_tensorcnt) and shared by all 8 waves.
// Each wave reuses one A fragment across 4 WMMAs (4 N-tiles).
// grid = (N/64, 1, Z).  K1, K2 multiples of 32.
// ---------------------------------------------------------------------------
__global__ __launch_bounds__(256) void wmma_gemm_block_kernel(
    const __bf16* __restrict__ A1, long long a1_zs, int lda1, int K1,
    const __bf16* __restrict__ A2, long long a2_zs, int lda2, int K2,
    const __bf16* __restrict__ W1, long long w1_zs, int ldw1,
    const __bf16* __restrict__ W2, long long w2_zs, int ldw2,
    const float*  __restrict__ bias, long long bias_zs,
    float*        __restrict__ D, long long d_zs, int ldd)
{
  __shared__ __bf16 wtile[2][64][WROW];

  const int tid   = threadIdx.x;
  const int lane  = tid & 31;
  const int wv    = tid >> 5;                 // 0..7 -> M tile
  const int z     = blockIdx.z;
  const int nbase = blockIdx.x * 64;

  const __bf16* A1z = A1 + (long long)z * a1_zs;
  const __bf16* A2z = A2 + (long long)z * a2_zs;
  const __bf16* W1z = W1 + (long long)z * w1_zs;
  const __bf16* W2z = W2 + (long long)z * w2_zs;

  const int c1  = K1 >> 5;
  const int nch = c1 + (K2 >> 5);

  // Stage one 64x32 bf16 weight chunk into wtile[bi].
  auto stage = [&](int c, int bi) {
    const __bf16* wsrc;
    int ldw;
    if (c < c1) { wsrc = W1z + (long long)nbase * ldw1 + (c << 5);        ldw = ldw1; }
    else        { wsrc = W2z + (long long)nbase * ldw2 + ((c - c1) << 5); ldw = ldw2; }
#ifdef USE_TDM
    if (wv == 0) {
      unsigned long long ga = (unsigned long long)wsrc;
      unsigned lds = (unsigned)(unsigned long long)
          (__attribute__((address_space(3))) char*)&wtile[bi][0][0];
      // D# group 0: count=1 | lds_addr | global_addr | type=2
      u32x4 g0;
      g0[0] = 1u;
      g0[1] = lds;
      g0[2] = (unsigned)ga;
      g0[3] = (unsigned)((ga >> 32) & 0x01FFFFFFu) | (2u << 30);
      // D# group 1: data_size=2B, pad_enable, pad every 16 DW by 4 DW (row 64B -> 80B),
      // tensor_dim0=32, tensor_dim1=64, tile_dim0=32, tile_dim1=64, dim0_stride=ldw
      i32x8 g1;
      g1[0] = (int)((1u << 16) | (1u << 20) | (3u << 22) | (3u << 25));
      g1[1] = (int)(32u << 16);
      g1[2] = (int)(64u << 16);
      g1[3] = (int)(32u << 16);
      g1[4] = 64;
      g1[5] = ldw;
      g1[6] = 0;
      g1[7] = 0;
      i32x4 zz4 = {0, 0, 0, 0};
#if defined(__clang_major__) && (__clang_major__ >= 23)
      i32x8 zz8 = {0, 0, 0, 0, 0, 0, 0, 0};
      __builtin_amdgcn_tensor_load_to_lds(g0, g1, zz4, zz4, zz8, 0);   // 6-arg form
#else
      __builtin_amdgcn_tensor_load_to_lds(g0, g1, zz4, zz4, 0);        // 5-arg form (ROCm 7.2)
#endif
    }
#else
    // Fallback: cooperative synchronous staging (load + ds_store).
    int r = tid >> 2, q = tid & 3;
    const uint4 v = *reinterpret_cast<const uint4*>(wsrc + (long long)r * ldw + (q << 3));
    *reinterpret_cast<uint4*>(&wtile[bi][r][q << 3]) = v;
#endif
  };

  f32x8 acc[4];
#pragma unroll
  for (int j = 0; j < 4; ++j) acc[j] = f32x8{0.f, 0.f, 0.f, 0.f, 0.f, 0.f, 0.f, 0.f};

  const int mrow  = wv * 16 + (lane & 15);
  const int ahalf = (lane >> 4) ? 16 : 0;
  const int brow  = lane & 15;
  const int bk    = (lane >> 4) ? 16 : 0;

  stage(0, 0);

  for (int c = 0; c < nch; ++c) {
    if (c + 1 < nch) stage(c + 1, (c + 1) & 1);
#ifdef USE_TDM
    if (wv == 0) {
      if (c + 1 < nch) __builtin_amdgcn_s_wait_tensorcnt((short)1);  // chunk c landed
      else             __builtin_amdgcn_s_wait_tensorcnt((short)0);
    }
#endif
    __syncthreads();   // staging of chunk c visible to all waves

    const __bf16* abase;
    if (c < c1) abase = A1z + (long long)mrow * lda1 + (c << 5);
    else        abase = A2z + (long long)mrow * lda2 + ((c - c1) << 5);
    bf16x8 alo = *reinterpret_cast<const bf16x8*>(abase + ahalf);
    bf16x8 ahi = *reinterpret_cast<const bf16x8*>(abase + ahalf + 8);
    bf16x16 a;
#pragma unroll
    for (int i = 0; i < 8; ++i) { a[i] = alo[i]; a[i + 8] = ahi[i]; }

    const __bf16* wl = &wtile[c & 1][0][0];
#pragma unroll
    for (int j = 0; j < 4; ++j) {
      const __bf16* bp = wl + (j * 16 + brow) * WROW + bk;
      bf16x8 blo = *reinterpret_cast<const bf16x8*>(bp);
      bf16x8 bhi = *reinterpret_cast<const bf16x8*>(bp + 8);
      bf16x16 b;
#pragma unroll
      for (int i = 0; i < 8; ++i) { b[i] = blo[i]; b[i + 8] = bhi[i]; }
      acc[j] = __builtin_amdgcn_wmma_f32_16x16x32_bf16(false, a, false, b,
                                                       (short)0, acc[j], false, false);
    }
    __syncthreads();   // all waves done with wtile[c&1] before it is restaged
  }

  const float* bz = bias + (long long)z * bias_zs;
  float* dz = D + (long long)z * d_zs;
  const int mbase = wv * 16 + ((lane >> 4) << 3);
#pragma unroll
  for (int j = 0; j < 4; ++j) {
    const int ncol = nbase + j * 16 + (lane & 15);
    const float bn = bz[ncol];
#pragma unroll
    for (int r = 0; r < 8; ++r)
      dz[(long long)(mbase + r) * ldd + ncol] = acc[j][r] + bn;
  }
}

// ---------------------------------------------------------------------------
// Single-wave WMMA GEMM (kept for the small decoder-logit GEMM).
// grid = (N/16, M/16, Z), block = 32.
// ---------------------------------------------------------------------------
__global__ __launch_bounds__(32) void wmma_gemm_kernel(
    const __bf16* __restrict__ A1, long long a1_zs, int lda1, int K1,
    const __bf16* __restrict__ A2, long long a2_zs, int lda2, int K2,
    const __bf16* __restrict__ W1, long long w1_zs, int ldw1,
    const __bf16* __restrict__ W2, long long w2_zs, int ldw2,
    const float*  __restrict__ bias, long long bias_zs,
    float*        __restrict__ D, long long d_zs, int ldd)
{
  const int lane = threadIdx.x;
  const int z    = blockIdx.z;
  const int tn   = blockIdx.x * 16;
  const int tm   = blockIdx.y * 16;

  const int mrow  = tm + (lane & 15);
  const int ncol  = tn + (lane & 15);
  const int ahalf = (lane >> 4) ? 16 : 0;
  const int bhalf = (lane >> 4) ? 16 : 0;

  f32x8 acc = {0.f, 0.f, 0.f, 0.f, 0.f, 0.f, 0.f, 0.f};

  {
    const __bf16* arow = A1 + (long long)z * a1_zs + (long long)mrow * lda1;
    const __bf16* wrow = W1 + (long long)z * w1_zs + (long long)ncol * ldw1;
    for (int k0 = 0; k0 < K1; k0 += 32) {
      bf16x8 lo = *reinterpret_cast<const bf16x8*>(arow + k0 + ahalf);
      bf16x8 hi = *reinterpret_cast<const bf16x8*>(arow + k0 + ahalf + 8);
      bf16x16 a;
#pragma unroll
      for (int i = 0; i < 8; ++i) { a[i] = lo[i]; a[i + 8] = hi[i]; }
      bf16x16 b = *reinterpret_cast<const bf16x16*>(wrow + k0 + bhalf);
      acc = __builtin_amdgcn_wmma_f32_16x16x32_bf16(false, a, false, b,
                                                    (short)0, acc, false, false);
    }
  }
  if (K2 > 0) {
    const __bf16* arow = A2 + (long long)z * a2_zs + (long long)mrow * lda2;
    const __bf16* wrow = W2 + (long long)z * w2_zs + (long long)ncol * ldw2;
    for (int k0 = 0; k0 < K2; k0 += 32) {
      bf16x8 lo = *reinterpret_cast<const bf16x8*>(arow + k0 + ahalf);
      bf16x8 hi = *reinterpret_cast<const bf16x8*>(arow + k0 + ahalf + 8);
      bf16x16 a;
#pragma unroll
      for (int i = 0; i < 8; ++i) { a[i] = lo[i]; a[i + 8] = hi[i]; }
      bf16x16 b = *reinterpret_cast<const bf16x16*>(wrow + k0 + bhalf);
      acc = __builtin_amdgcn_wmma_f32_16x16x32_bf16(false, a, false, b,
                                                    (short)0, acc, false, false);
    }
  }

  const float bn = bias ? bias[(long long)z * bias_zs + ncol] : 0.f;
  float* drow = D + (long long)z * d_zs;
  const int mbase = tm + ((lane >> 4) << 3);
#pragma unroll
  for (int r = 0; r < 8; ++r)
    drow[(long long)(mbase + r) * ldd + ncol] = acc[r] + bn;
}

// ---------------------------------------------------------------------------
// Embedding gather (+bf16 convert, zero-pad 300->320). dst layout (t, b, E_PAD)
// ---------------------------------------------------------------------------
__global__ void embed_gather_kernel(const int* __restrict__ X,
                                    const float* __restrict__ emb,
                                    __bf16* __restrict__ dst)
{
  int tb = blockIdx.x;            // t*B + b
  int b  = tb & (B_ - 1);
  int t  = tb >> 7;
  int j  = threadIdx.x;           // 0..319
  int vi = X[b * T_ + t];
  float val = (j < E_) ? emb[(long long)vi * E_ + j] : 0.f;
  dst[(long long)tb * E_PAD + j] = f2bf(val);
}

// fp32 -> bf16 convert with optional column zero-padding
__global__ void cvt_pad_kernel(const float* __restrict__ src, __bf16* __restrict__ dst,
                               int srcCols, int dstCols, long long total)
{
  long long i = (long long)blockIdx.x * blockDim.x + threadIdx.x;
  long long stride = (long long)gridDim.x * blockDim.x;
  for (; i < total; i += stride) {
    long long r = i / dstCols;
    int c = (int)(i - r * dstCols);
    float v = (c < srcCols) ? src[r * srcCols + c] : 0.f;
    dst[i] = f2bf(v);
  }
}

// ---------------------------------------------------------------------------
// LSTM cell update. g layout (z, b, 4H) gates [i|f|g|o].
// ---------------------------------------------------------------------------
__global__ void lstm_cell_kernel(const float* __restrict__ g, long long g_zs,
                                 float* __restrict__ c, float* __restrict__ h, long long hc_zs,
                                 __bf16* __restrict__ hbf, long long hbf_zs,
                                 __bf16* __restrict__ outbf, long long out_off0, long long out_off1,
                                 float* __restrict__ hmax)
{
  int z   = blockIdx.z;
  int idx = blockIdx.x * blockDim.x + threadIdx.x;  // 0..B*H-1
  int b   = idx >> 9;
  int j   = idx & (H_ - 1);
  const float* gz = g + (long long)z * g_zs + (long long)b * (4 * H_);
  float i_ = sigf(gz[j]);
  float f_ = sigf(gz[H_ + j]);
  float g_ = tanhf(gz[2 * H_ + j]);
  float o_ = sigf(gz[3 * H_ + j]);
  float* cz = c + (long long)z * hc_zs;
  float cn = f_ * cz[idx] + i_ * g_;
  cz[idx] = cn;
  float hn = o_ * tanhf(cn);
  h[(long long)z * hc_zs + idx] = hn;
  hbf[(long long)z * hbf_zs + idx] = f2bf(hn);
  if (outbf) {
    long long off = z ? out_off1 : out_off0;
    outbf[off + (long long)b * (2 * H_) + z * H_ + j] = f2bf(hn);
  }
  if (hmax) hmax[idx] = fmaxf(hmax[idx], hn);
}

// ---------------------------------------------------------------------------
// Norm stats over (T*B) rows of 1024 channels, deterministic two-stage.
// ---------------------------------------------------------------------------
__global__ __launch_bounds__(256) void bn_stats_kernel(const __bf16* __restrict__ x,
                                                       float* __restrict__ ps,
                                                       float* __restrict__ pq)
{
  int ch = blockIdx.x * 256 + threadIdx.x;       // 0..1023
  int r0 = blockIdx.y * 512;                     // 64 chunks of 512 rows
  float s = 0.f, s2 = 0.f;
  for (int r = r0; r < r0 + 512; ++r) {
    float v = bf2f(x[(long long)r * 1024 + ch]);
    s += v; s2 += v * v;
  }
  ps[(long long)blockIdx.y * 1024 + ch] = s;
  pq[(long long)blockIdx.y * 1024 + ch] = s2;
}

__global__ void bn_finalize_kernel(const float* __restrict__ ps, const float* __restrict__ pq,
                                   float* __restrict__ mu, float* __restrict__ rsig)
{
  int ch = blockIdx.x * blockDim.x + threadIdx.x;   // 0..1023
  float s = 0.f, s2 = 0.f;
  for (int r = 0; r < 64; ++r) { s += ps[(long long)r * 1024 + ch]; s2 += pq[(long long)r * 1024 + ch]; }
  const float n = (float)(T_ * B_);
  float m = s / n;
  float var = s2 / n - m * m;
  mu[ch] = m;
  rsig[ch] = rsqrtf(var + 1e-5f);
}

__global__ void bn_apply_kernel(const __bf16* __restrict__ src, __bf16* __restrict__ dst,
                                const float* __restrict__ mu, const float* __restrict__ rsig,
                                const float* __restrict__ gamma, const float* __restrict__ beta)
{
  long long total = (long long)T_ * B_ * 1024;
  long long i = (long long)blockIdx.x * 256 + threadIdx.x;
  long long stride = (long long)gridDim.x * 256;
  for (; i < total; i += stride) {
    int ch = (int)(i & 1023);
    float v = bf2f(src[i]);
    dst[i] = f2bf((v - mu[ch]) * rsig[ch] * gamma[ch] + beta[ch]);
  }
}

// Row softmax over 256 elems, in place. grid = rows, block = 256.
__global__ __launch_bounds__(256) void softmax256_kernel(float* __restrict__ x)
{
  __shared__ float buf[256];
  int r = blockIdx.x, tid = threadIdx.x;
  float v = x[(long long)r * 256 + tid];
  buf[tid] = v; __syncthreads();
  for (int off = 128; off > 0; off >>= 1) { if (tid < off) buf[tid] = fmaxf(buf[tid], buf[tid + off]); __syncthreads(); }
  float m = buf[0]; __syncthreads();
  float e = expf(v - m);
  buf[tid] = e; __syncthreads();
  for (int off = 128; off > 0; off >>= 1) { if (tid < off) buf[tid] += buf[tid + off]; __syncthreads(); }
  x[(long long)r * 256 + tid] = e / buf[0];
}

// Batched attention context: v[b, h*1024+d] = sum_l att[h,b,l]*x1n[l,b,d]
__global__ __launch_bounds__(256) void att_context_kernel(const float* __restrict__ att,
                                                          const __bf16* __restrict__ x1n,
                                                          __bf16* __restrict__ vbf)
{
  __shared__ float sa[3][256];
  int b = blockIdx.y, tid = threadIdx.x;
  for (int hh = 0; hh < 3; ++hh)
    sa[hh][tid] = att[((long long)hh * B_ + b) * 256 + tid];
  __syncthreads();
  int d = blockIdx.x * 256 + tid;                  // 0..1023
  float a0 = 0.f, a1 = 0.f, a2 = 0.f;
  const __bf16* base = x1n + (long long)b * 1024 + d;
  for (int l = 0; l < 256; ++l) {
    float xv = bf2f(base[(long long)l * B_ * 1024]);
    a0 += sa[0][l] * xv; a1 += sa[1][l] * xv; a2 += sa[2][l] * xv;
  }
  __bf16* vb = vbf + (long long)b * 3072;
  vb[d] = f2bf(a0); vb[1024 + d] = f2bf(a1); vb[2048 + d] = f2bf(a2);
}

// Small fp32 GEMM (heads): D[m,n] = act(sum_k A[m,k]*W[n,k] + bias[n])
__global__ void head_gemm_kernel(const float* __restrict__ A, int lda,
                                 const float* __restrict__ W, const float* __restrict__ bias,
                                 float* __restrict__ D, int M, int N, int K, int leaky)
{
  int i = blockIdx.x * blockDim.x + threadIdx.x;
  if (i >= M * N) return;
  int m = i / N, n = i - m * N;
  float s = bias[n];
  const float* a = A + (long long)m * lda;
  const float* w = W + (long long)n * K;
  for (int k = 0; k < K; ++k) s += a[k] * w[k];
  if (leaky) s = s > 0.f ? s : NEG_ * s;
  D[i] = s;
}

__global__ void fill_f32_kernel(float* __restrict__ p, float v, long long n)
{
  long long i = (long long)blockIdx.x * 256 + threadIdx.x;
  if (i < n) p[i] = v;
}

// Final fusion head: feats -> r -> attention over 4 -> s -> (y1, y2). block=64, grid=B.
__global__ __launch_bounds__(64) void final_head_kernel(
    const float* __restrict__ yfeat, const float* __restrict__ Fe,
    const float* __restrict__ fe2, const float* __restrict__ fe3,
    const float* __restrict__ WR, const float* __restrict__ bR, const float* __restrict__ WA,
    const float* __restrict__ Wt1, const float* __restrict__ bt1,
    const float* __restrict__ Wt2, const float* __restrict__ bt2,
    float* __restrict__ out)
{
  __shared__ float feats[4][FEAT_];
  __shared__ float rmat[4][64];
  __shared__ float red[64];
  __shared__ float aw[4];
  __shared__ float svec[64];
  int b = blockIdx.x, n = threadIdx.x;
  for (int i = n; i < FEAT_; i += 64) {
    feats[0][i] = yfeat[b * FEAT_ + i];
    feats[1][i] = Fe[(long long)b * 929 + i];
    feats[2][i] = fe2[b * FEAT_ + i];
    feats[3][i] = fe3[b * FEAT_ + i];
  }
  __syncthreads();
  for (int k = 0; k < 4; ++k) {
    float s = bR[n];
    for (int d = 0; d < FEAT_; ++d) s += feats[k][d] * WR[n * FEAT_ + d];
    rmat[k][n] = s > 0.f ? s : NEG_ * s;
  }
  __syncthreads();
  for (int k = 0; k < 4; ++k) {
    red[n] = tanhf(rmat[k][n]) * WA[n];
    __syncthreads();
    for (int off = 32; off > 0; off >>= 1) { if (n < off) red[n] += red[n + off]; __syncthreads(); }
    if (n == 0) aw[k] = red[0];
    __syncthreads();
  }
  if (n == 0) {
    float m = fmaxf(fmaxf(aw[0], aw[1]), fmaxf(aw[2], aw[3]));
    float e0 = expf(aw[0] - m), e1 = expf(aw[1] - m), e2 = expf(aw[2] - m), e3 = expf(aw[3] - m);
    float s = e0 + e1 + e2 + e3;
    aw[0] = e0 / s; aw[1] = e1 / s; aw[2] = e2 / s; aw[3] = e3 / s;
  }
  __syncthreads();
  float sv = 0.f;
  for (int k = 0; k < 4; ++k) sv += aw[k] * rmat[k][n];
  svec[n] = fmaxf(sv, 0.f);
  __syncthreads();
  if (n < 2) {
    float y1 = bt1[n], y2 = bt2[n];
    for (int d = 0; d < 64; ++d) { y1 += svec[d] * Wt1[n * 64 + d]; y2 += svec[d] * Wt2[n * 64 + d]; }
    out[b * 2 + n]       = y1;
    out[256 + b * 2 + n] = y2;
  }
}

// ---------------------------------------------------------------------------
extern "C" void kernel_launch(void* const* d_in, const int* in_sizes, int n_in,
                              void* d_out, int out_size, void* d_ws, size_t ws_size,
                              hipStream_t stream)
{
  (void)in_sizes; (void)n_in; (void)out_size; (void)ws_size;

  const int*   X       = (const int*)  d_in[0];
  const float* Fe      = (const float*)d_in[1];
  const float* emb     = (const float*)d_in[2];
  const float* Wih0    = (const float*)d_in[3];
  const float* Whh0    = (const float*)d_in[4];
  const float* b0      = (const float*)d_in[5];
  const float* Wih1    = (const float*)d_in[6];
  const float* Whh1    = (const float*)d_in[7];
  const float* b1      = (const float*)d_in[8];
  const float* gamma   = (const float*)d_in[9];
  const float* beta    = (const float*)d_in[10];
  const float* Wa1     = (const float*)d_in[11];
  const float* ba1     = (const float*)d_in[12];
  const float* Wa2     = (const float*)d_in[13];
  const float* ba2     = (const float*)d_in[14];
  const float* Wa3     = (const float*)d_in[15];
  const float* ba3     = (const float*)d_in[16];
  const float* Wc_ih   = (const float*)d_in[17];
  const float* Wc_hh   = (const float*)d_in[18];
  const float* bc      = (const float*)d_in[19];
  const float* Wdense1 = (const float*)d_in[20];
  const float* bdense1 = (const float*)d_in[21];
  const float* W3      = (const float*)d_in[22];
  const float* b3      = (const float*)d_in[23];
  const float* W4a     = (const float*)d_in[24];
  const float* b4a     = (const float*)d_in[25];
  const float* W4b     = (const float*)d_in[26];
  const float* b4b     = (const float*)d_in[27];
  const float* WR      = (const float*)d_in[28];
  const float* bR      = (const float*)d_in[29];
  const float* WA      = (const float*)d_in[30];
  const float* Wt1     = (const float*)d_in[31];
  const float* bt1     = (const float*)d_in[32];
  const float* Wt2     = (const float*)d_in[33];
  const float* bt2     = (const float*)d_in[34];

  // ---- workspace layout ----
  char* ws = (char*)d_ws;
  size_t off = 0;
  auto alloc = [&](size_t bytes) -> char* {
    char* p = ws + off;
    off += (bytes + 255) & ~(size_t)255;
    return p;
  };
  __bf16* XembBF = (__bf16*)alloc((size_t)T_ * B_ * E_PAD * 2);
  __bf16* W0ih   = (__bf16*)alloc((size_t)2 * 2048 * E_PAD * 2);
  __bf16* W0hh   = (__bf16*)alloc((size_t)2 * 2048 * 512 * 2);
  __bf16* W1ih   = (__bf16*)alloc((size_t)2 * 2048 * 1024 * 2);
  __bf16* W1hh   = (__bf16*)alloc((size_t)2 * 2048 * 512 * 2);
  __bf16* WaBF   = (__bf16*)alloc((size_t)3 * 256 * 512 * 2);
  float*  baCat  = (float*) alloc((size_t)3 * 256 * 4);
  __bf16* WcihBF = (__bf16*)alloc((size_t)2048 * 3072 * 2);
  __bf16* WchhBF = (__bf16*)alloc((size_t)2048 * 512 * 2);
  __bf16* bufA   = (__bf16*)alloc((size_t)T_ * B_ * 1024 * 2); // L0 out, then x1 normalized
  __bf16* bufB   = (__bf16*)alloc((size_t)T_ * B_ * 1024 * 2); // L1 out
  float*  henc   = (float*) alloc((size_t)2 * B_ * H_ * 4);
  float*  cenc   = (float*) alloc((size_t)2 * B_ * H_ * 4);
  __bf16* hencBF = (__bf16*)alloc((size_t)2 * B_ * H_ * 2);
  float*  genc   = (float*) alloc((size_t)2 * B_ * 2048 * 4);
  float*  statPS = (float*) alloc((size_t)64 * 1024 * 4);
  float*  statPQ = (float*) alloc((size_t)64 * 1024 * 4);
  float*  muB    = (float*) alloc(1024 * 4);
  float*  rsB    = (float*) alloc(1024 * 4);
  float*  hdec   = (float*) alloc((size_t)B_ * H_ * 4);
  float*  cdec   = (float*) alloc((size_t)B_ * H_ * 4);
  __bf16* hdecBF = (__bf16*)alloc((size_t)B_ * H_ * 2);
  float*  attB   = (float*) alloc((size_t)3 * B_ * 256 * 4);
  __bf16* vBF    = (__bf16*)alloc((size_t)B_ * 3072 * 2);
  float*  gdec   = (float*) alloc((size_t)B_ * 2048 * 4);
  float*  hsmax  = (float*) alloc((size_t)B_ * H_ * 4);
  float*  yfeat  = (float*) alloc((size_t)B_ * FEAT_ * 4);
  float*  fe2b   = (float*) alloc((size_t)B_ * FEAT_ * 4);
  float*  fe3a   = (float*) alloc((size_t)B_ * 192 * 4);
  float*  fe3b   = (float*) alloc((size_t)B_ * FEAT_ * 4);

  // ---- weight conversions to bf16 (+K padding) ----
  cvt_pad_kernel<<<4096, 256, 0, stream>>>(Wih0, W0ih, 300, 320, (long long)4096 * 320);
  cvt_pad_kernel<<<4096, 256, 0, stream>>>(Whh0, W0hh, 512, 512, (long long)4096 * 512);
  cvt_pad_kernel<<<4096, 256, 0, stream>>>(Wih1, W1ih, 1024, 1024, (long long)4096 * 1024);
  cvt_pad_kernel<<<4096, 256, 0, stream>>>(Whh1, W1hh, 512, 512, (long long)4096 * 512);
  cvt_pad_kernel<<<512, 256, 0, stream>>>(Wa1, WaBF,                 512, 512, (long long)256 * 512);
  cvt_pad_kernel<<<512, 256, 0, stream>>>(Wa2, WaBF + 256 * 512,     512, 512, (long long)256 * 512);
  cvt_pad_kernel<<<512, 256, 0, stream>>>(Wa3, WaBF + 2 * 256 * 512, 512, 512, (long long)256 * 512);
  cvt_pad_kernel<<<4096, 256, 0, stream>>>(Wc_ih, WcihBF, 3072, 3072, (long long)2048 * 3072);
  cvt_pad_kernel<<<2048, 256, 0, stream>>>(Wc_hh, WchhBF, 512, 512, (long long)2048 * 512);
  (void)hipMemcpyAsync(baCat,       ba1, 256 * 4, hipMemcpyDeviceToDevice, stream);
  (void)hipMemcpyAsync(baCat + 256, ba2, 256 * 4, hipMemcpyDeviceToDevice, stream);
  (void)hipMemcpyAsync(baCat + 512, ba3, 256 * 4, hipMemcpyDeviceToDevice, stream);

  // ---- embedding gather ----
  embed_gather_kernel<<<T_ * B_, E_PAD, 0, stream>>>(X, emb, XembBF);

  const dim3 gBlk(2048 / 64, 1, 2);       // blocked GEMM, both directions
  const dim3 gCell(B_ * H_ / 256, 1, 2);

  // ---- BiLSTM layer 0 ----
  (void)hipMemsetAsync(henc,   0, (size_t)2 * B_ * H_ * 4, stream);
  (void)hipMemsetAsync(cenc,   0, (size_t)2 * B_ * H_ * 4, stream);
  (void)hipMemsetAsync(hencBF, 0, (size_t)2 * B_ * H_ * 2, stream);
  for (int t = 0; t < T_; ++t) {
    long long a1zs = (long long)(T_ - 1 - 2 * t) * B_ * E_PAD; // dir1 reads time T-1-t
    wmma_gemm_block_kernel<<<gBlk, 256, 0, stream>>>(
        XembBF + (long long)t * B_ * E_PAD, a1zs, E_PAD, E_PAD,
        hencBF, (long long)B_ * H_, H_, H_,
        W0ih, (long long)2048 * E_PAD, E_PAD,
        W0hh, (long long)2048 * 512, 512,
        b0, 2048,
        genc, (long long)B_ * 2048, 2048);
    lstm_cell_kernel<<<gCell, 256, 0, stream>>>(
        genc, (long long)B_ * 2048,
        cenc, henc, (long long)B_ * H_,
        hencBF, (long long)B_ * H_,
        bufA, (long long)t * B_ * 1024, (long long)(T_ - 1 - t) * B_ * 1024,
        nullptr);
  }

  // ---- BiLSTM layer 1 ----
  (void)hipMemsetAsync(henc,   0, (size_t)2 * B_ * H_ * 4, stream);
  (void)hipMemsetAsync(cenc,   0, (size_t)2 * B_ * H_ * 4, stream);
  (void)hipMemsetAsync(hencBF, 0, (size_t)2 * B_ * H_ * 2, stream);
  for (int t = 0; t < T_; ++t) {
    long long a1zs = (long long)(T_ - 1 - 2 * t) * B_ * 1024;
    wmma_gemm_block_kernel<<<gBlk, 256, 0, stream>>>(
        bufA + (long long)t * B_ * 1024, a1zs, 1024, 1024,
        hencBF, (long long)B_ * H_, H_, H_,
        W1ih, (long long)2048 * 1024, 1024,
        W1hh, (long long)2048 * 512, 512,
        b1, 2048,
        genc, (long long)B_ * 2048, 2048);
    lstm_cell_kernel<<<gCell, 256, 0, stream>>>(
        genc, (long long)B_ * 2048,
        cenc, henc, (long long)B_ * H_,
        hencBF, (long long)B_ * H_,
        bufB, (long long)t * B_ * 1024, (long long)(T_ - 1 - t) * B_ * 1024,
        nullptr);
  }

  // ---- normalization over (B,T) per channel; normalized x1 -> bufA (bf16) ----
  bn_stats_kernel<<<dim3(4, 64), 256, 0, stream>>>(bufB, statPS, statPQ);
  bn_finalize_kernel<<<4, 256, 0, stream>>>(statPS, statPQ, muB, rsB);
  bn_apply_kernel<<<8192, 256, 0, stream>>>(bufB, bufA, muB, rsB, gamma, beta);

  // ---- attention-LSTM decoder ----
  (void)hipMemsetAsync(hdec,   0, (size_t)B_ * H_ * 4, stream);
  (void)hipMemsetAsync(cdec,   0, (size_t)B_ * H_ * 4, stream);
  (void)hipMemsetAsync(hdecBF, 0, (size_t)B_ * H_ * 2, stream);
  fill_f32_kernel<<<(B_ * H_ + 255) / 256, 256, 0, stream>>>(hsmax, -3.0e38f, (long long)B_ * H_);

  const dim3 gLog(256 / 16, B_ / 16, 3);
  const dim3 gDec(2048 / 64, 1, 1);
  for (int t = 0; t < T_; ++t) {
    // logits for 3 heads: (128x512)@(512x256)^T
    wmma_gemm_kernel<<<gLog, 32, 0, stream>>>(
        hdecBF, 0, H_, H_,
        nullptr, 0, 0, 0,
        WaBF, (long long)256 * 512, 512,
        nullptr, 0, 0,
        baCat, 256,
        attB, (long long)B_ * 256, 256);
    softmax256_kernel<<<3 * B_, 256, 0, stream>>>(attB);
    att_context_kernel<<<dim3(4, B_), 256, 0, stream>>>(attB, bufA, vBF);
    // g = v @ Wc_ih^T + h @ Wc_hh^T + bc
    wmma_gemm_block_kernel<<<gDec, 256, 0, stream>>>(
        vBF, 0, 3072, 3072,
        hdecBF, 0, H_, H_,
        WcihBF, 0, 3072,
        WchhBF, 0, 512,
        bc, 0,
        gdec, 0, 2048);
    lstm_cell_kernel<<<dim3(B_ * H_ / 256, 1, 1), 256, 0, stream>>>(
        gdec, 0, cdec, hdec, 0, hdecBF, 0,
        nullptr, 0, 0, hsmax);
  }

  // ---- feature heads (tiny fp32 GEMMs) ----
  head_gemm_kernel<<<(B_ * FEAT_ + 255) / 256, 256, 0, stream>>>(hsmax, 512, Wdense1, bdense1, yfeat, B_, FEAT_, 512, 1);
  head_gemm_kernel<<<(B_ * FEAT_ + 255) / 256, 256, 0, stream>>>(Fe + 61, 929, W3, b3, fe2b, B_, FEAT_, 100, 1);
  head_gemm_kernel<<<(B_ * 192 + 255) / 256, 256, 0, stream>>>(Fe + 161, 929, W4a, b4a, fe3a, B_, 192, 768, 1);
  head_gemm_kernel<<<(B_ * FEAT_ + 255) / 256, 256, 0, stream>>>(fe3a, 192, W4b, b4b, fe3b, B_, FEAT_, 192, 1);

  // ---- final fusion head -> (y1, y2) ----
  final_head_kernel<<<B_, 64, 0, stream>>>(yfeat, Fe, fe2b, fe3b, WR, bR, WA,
                                           Wt1, bt1, Wt2, bt2, (float*)d_out);
}
